// Bayesian_MultiHeadSelfAttention_Intermediate_27745488732905
// MI455X (gfx1250) — matrix operand order repaired and checked by reading, not compiled, not run on hardware
//
#include <hip/hip_runtime.h>

#define E_DIM 768
#define SEQ   1024
#define BATCH 4
#define HEADS 12
#define HDIM  64
#define ROWS  (BATCH*SEQ)        // 4096
#define BH    (BATCH*HEADS)      // 48
#define HELEMS ((size_t)BH*SEQ*HDIM)  // 3145728 == ROWS*E_DIM
#define SP    1032               // f32 LDS row pitch, breaks bank conflicts
#define SPB   1040               // bf16 LDS row pitch (16B-aligned rows, bank-staggered)

typedef __bf16 bf16;
typedef __attribute__((ext_vector_type(16))) __bf16 v16bf;
typedef __attribute__((ext_vector_type(8)))  __bf16 v8bf;
typedef __attribute__((ext_vector_type(8)))  float  v8f;

__device__ __forceinline__ float softplusf(float x) {
  return (x > 20.0f) ? x : log1pf(__expf(x));
}

__device__ __forceinline__ v8f wmma_bf16(v16bf a, v16bf b, v8f c) {
  // D = A(16x32) * B(32x16) + C, f32 accumulate
  return __builtin_amdgcn_wmma_f32_16x16x32_bf16(false, a, false, b, (short)0, c, false, false);
}

// A fragment 16x32 bf16 from row-major [*, ld]; base pre-offset to (m0, k0).
// ISA layout: lanes 0-15 row M=lane, K 0-7 in v0-3, K 16-23 in v4-7; lanes 16-31: K 8-15 / 24-31.
// Works for both global and LDS-derived pointers (addrspace inferred -> ds_load for shared).
__device__ __forceinline__ v16bf load_a(const bf16* base, int ld, int lane) {
  int m = lane & 15, hi = lane >> 4;
  const bf16* p = base + m * ld + hi * 8;
  v16bf a;
  *((v8bf*)&a)       = *((const v8bf*)p);        // K = hi*8 .. hi*8+7
  *(((v8bf*)&a) + 1) = *((const v8bf*)(p + 16)); // K = 16+hi*8 ..
  return a;
}

// B fragment 32x16 (KxN); source stored "transposed": row n holds its K data contiguously.
// ISA layout: lanes 0-15 col N=lane, K 0-7 in v0-3, K 16-23 in v4-7; lanes 16-31: K 8-15 / 24-31.
__device__ __forceinline__ v16bf load_b(const bf16* base, int ld, int lane) {
  int n = lane & 15, hi = lane >> 4;
  const bf16* p = base + n * ld + hi * 8;
  v16bf b;
  *((v8bf*)&b)       = *((const v8bf*)p);
  *(((v8bf*)&b) + 1) = *((const v8bf*)(p + 16));
  return b;
}

// ---------------- prep kernels ----------------

__global__ void prep_weights_kernel(const float* __restrict__ w, const float* __restrict__ wsig,
                                    bf16* __restrict__ wt, bf16* __restrict__ w2t,
                                    float* __restrict__ spw) {
  int idx = blockIdx.x * blockDim.x + threadIdx.x;
  if (idx < E_DIM * E_DIM) {
    int n = idx / E_DIM, k = idx - n * E_DIM;
    float v = w[k * E_DIM + n];           // transpose: Wt[n][k] = w[k][n]
    wt[idx]  = (bf16)v;
    w2t[idx] = (bf16)(v * v * (1.0f / (float)E_DIM));  // (w/sqrt(Din))^2
  }
  if (idx < E_DIM) spw[idx] = softplusf(wsig[idx]);
}

__global__ void prep_inputs_kernel(const float* __restrict__ mu, const float* __restrict__ sg,
                                   bf16* __restrict__ mub, bf16* __restrict__ sgb,
                                   float* __restrict__ xx, float* __restrict__ ss) {
  __shared__ float r1[256], r2[256];
  int row = blockIdx.x, tid = threadIdx.x;
  const float* mr = mu + (size_t)row * E_DIM;
  const float* sr = sg + (size_t)row * E_DIM;
  float a = 0.0f, b = 0.0f;
  for (int j = tid; j < E_DIM; j += 256) {
    float m = mr[j], s = sr[j];
    mub[(size_t)row * E_DIM + j] = (bf16)m;
    sgb[(size_t)row * E_DIM + j] = (bf16)s;
    a += m * m; b += s;
  }
  r1[tid] = a; r2[tid] = b; __syncthreads();
  for (int o = 128; o; o >>= 1) {
    if (tid < o) { r1[tid] += r1[tid + o]; r2[tid] += r2[tid + o]; }
    __syncthreads();
  }
  if (tid == 0) {
    xx[row] = r1[0] * (1.0f / ((float)E_DIM * (float)E_DIM)); // sum((mu/Din)^2)
    ss[row] = r2[0];                                          // sum(Sigma_in)
  }
}

__global__ void cat_stats_kernel(const bf16* __restrict__ mub, const bf16* __restrict__ sgb,
                                 float* __restrict__ xx, float* __restrict__ ss) {
  __shared__ float r1[256], r2[256];
  int row = blockIdx.x, tid = threadIdx.x;
  float a = 0.0f, b = 0.0f;
  for (int j = tid; j < E_DIM; j += 256) {
    float m = (float)mub[(size_t)row * E_DIM + j];
    float s = (float)sgb[(size_t)row * E_DIM + j];
    a += m * m; b += s;
  }
  r1[tid] = a; r2[tid] = b; __syncthreads();
  for (int o = 128; o; o >>= 1) {
    if (tid < o) { r1[tid] += r1[tid + o]; r2[tid] += r2[tid + o]; }
    __syncthreads();
  }
  if (tid == 0) {
    xx[row] = r1[0] * (1.0f / ((float)E_DIM * (float)E_DIM));
    ss[row] = r2[0];
  }
}

// ---------------- linear_vdp (WMMA GEMM + variance epilogue) ----------------
// block = 256 threads = 8 waves; block covers 128 rows x 16 cols; one 16x16 WMMA tile per wave.

__global__ void vdp_linear_kernel(const bf16* __restrict__ Xmu, const bf16* __restrict__ Xsg,
                                  const bf16* __restrict__ Wt,  const bf16* __restrict__ W2t,
                                  const float* __restrict__ spw,
                                  const float* __restrict__ xx, const float* __restrict__ ss,
                                  bf16* __restrict__ omu_bf, bf16* __restrict__ osg_bf,
                                  float* __restrict__ omu_f, float* __restrict__ osg_f,
                                  int dbl) {
  int tid = threadIdx.x, lane = tid & 31, wave = tid >> 5;
  int n0 = blockIdx.x * 16;
  int m0 = blockIdx.y * 128 + wave * 16;
  const bf16* wb  = Wt  + (size_t)n0 * E_DIM;
  const bf16* w2b = W2t + (size_t)n0 * E_DIM;
  const bf16* xmu = Xmu + (size_t)m0 * E_DIM;
  const bf16* xsg = Xsg + (size_t)m0 * E_DIM;
  v8f accmu = {}; v8f accsg = {};
  for (int k0 = 0; k0 < E_DIM; k0 += 32) {
    if (k0 + 32 < E_DIM) {
      __builtin_prefetch(xmu + (lane & 15) * E_DIM + k0 + 32, 0, 1);
      __builtin_prefetch(xsg + (lane & 15) * E_DIM + k0 + 32, 0, 1);
    }
    v16bf a = load_a(xmu + k0, E_DIM, lane);
    v16bf b = load_b(wb  + k0, E_DIM, lane);
    accmu = wmma_bf16(a, b, accmu);
    v16bf a2 = load_a(xsg + k0, E_DIM, lane);
    v16bf b2 = load_b(w2b + k0, E_DIM, lane);
    accsg = wmma_bf16(a2, b2, accsg);
  }
  int n = lane & 15, rbase = (lane >> 4) * 8;
  int col = n0 + n;
  float sw = spw[col];
  #pragma unroll
  for (int r = 0; r < 8; ++r) {
    int row = m0 + rbase + r;
    float mu = accmu[r];
    // Sigma_1 + Sigma_2 + Sigma_3
    float sg = accsg[r] + xx[row] * sw + ss[row] * sw * (1.0f / (float)E_DIM);
    sg = softplusf(sg);
    if (dbl) sg = softplusf(sg);   // forward() applies softplus once more on the output layer
    size_t o = (size_t)row * E_DIM + col;
    if (omu_bf) { omu_bf[o] = (bf16)mu; osg_bf[o] = (bf16)sg; }
    else        { omu_f[o] = mu;        osg_f[o] = sg; }
  }
}

// ---------------- head packing ----------------

__global__ void pack_heads_kernel(const bf16* __restrict__ Qmu, const bf16* __restrict__ Qsg,
                                  const bf16* __restrict__ Kmu, const bf16* __restrict__ Ksg,
                                  const bf16* __restrict__ Vmu, const bf16* __restrict__ Vsg,
                                  bf16* __restrict__ hQmu, bf16* __restrict__ hQa,  bf16* __restrict__ hQsg,
                                  bf16* __restrict__ hKmu, bf16* __restrict__ hKmu2, bf16* __restrict__ hKsg,
                                  bf16* __restrict__ hVmuT, bf16* __restrict__ hVsgT, bf16* __restrict__ hVxT) {
  size_t idx = (size_t)blockIdx.x * 256 + threadIdx.x;
  if (idx >= HELEMS) return;
  int d  = (int)(idx & (HDIM - 1));
  int s  = (int)((idx >> 6) & (SEQ - 1));
  int bh = (int)(idx >> 16);
  int b = bh / HEADS, h = bh - b * HEADS;
  size_t src = ((size_t)(b * SEQ + s)) * E_DIM + h * HDIM + d;
  bf16 qm = Qmu[src], qs = Qsg[src];
  bf16 km = Kmu[src], ks = Ksg[src];
  bf16 vm = Vmu[src], vs = Vsg[src];
  float qmf = (float)qm, kmf = (float)km, vmf = (float)vm;
  hQmu[idx]  = qm;
  hQa[idx]   = (bf16)(qmf * qmf + (float)qs);   // folds 'a' + 'c' terms
  hQsg[idx]  = qs;
  hKmu[idx]  = km;
  hKmu2[idx] = (bf16)(kmf * kmf);               // 'b' term A operand
  hKsg[idx]  = ks;
  size_t tix = (size_t)bh * (SEQ * HDIM) + (size_t)d * SEQ + s;  // per-head transpose [bh][d][s]
  hVmuT[tix] = vm;
  hVsgT[tix] = vs;
  hVxT[tix]  = (bf16)(vmf * vmf + (float)vs);   // folds 'e' + 'f' terms
}

// ---------------- fused VDP attention ----------------
// One block = 4 waves (128 threads) handles one (b,h) and one 16-row q-tile.
// LDS: 16x1024 f32 mu-score + sigma-score rows (softmax precision) plus three
// pre-converted bf16 A-operand tiles (W, W^2, Sigma_weights) so the phase-2
// WMMA loop is pure ds_load + v_wmma with zero conversions.
// Total ~227KB -- only possible with CDNA5's 320KB/WGP LDS.

__global__ void attention_kernel(const bf16* __restrict__ hQmu, const bf16* __restrict__ hQa,
                                 const bf16* __restrict__ hQsg,
                                 const bf16* __restrict__ hKmu, const bf16* __restrict__ hKmu2,
                                 const bf16* __restrict__ hKsg,
                                 const bf16* __restrict__ hVmuT, const bf16* __restrict__ hVsgT,
                                 const bf16* __restrict__ hVxT,
                                 bf16* __restrict__ Cmu, bf16* __restrict__ Csg) {
  extern __shared__ float lds[];
  float* smu = lds;                 // [16][SP]  mu-score (f32)
  float* ssg = lds + 16 * SP;       // [16][SP]  sigma-score/d (f32)
  float* red = lds + 32 * SP;       // [128]     reduction scratch
  bf16* bW  = (bf16*)(red + 128);   // [16][SPB] softmax weights W        (bf16)
  bf16* bW2 = bW  + 16 * SPB;       // [16][SPB] W^2                      (bf16)
  bf16* bSw = bW2 + 16 * SPB;       // [16][SPB] Sigma_weights            (bf16)
  int tid = threadIdx.x, lane = tid & 31, wave = tid >> 5;
  int bh = blockIdx.x, q0 = blockIdx.y * 16;
  const size_t hoff = (size_t)bh * (SEQ * HDIM);
  const bf16* qmu = hQmu  + hoff + (size_t)q0 * HDIM;
  const bf16* qa  = hQa   + hoff + (size_t)q0 * HDIM;
  const bf16* km2 = hKmu2 + hoff + (size_t)q0 * HDIM;  // 'b' term: A rows are q-indexed

  // ---- phase 1: scores (3 WMMAs per 16x16 k-tile) ----
  for (int t = wave; t < SEQ / 16; t += 4) {
    int k0 = t * 16;
    const bf16* kmu = hKmu + hoff + (size_t)k0 * HDIM;
    const bf16* ksg = hKsg + hoff + (size_t)k0 * HDIM;
    const bf16* qsg = hQsg + hoff + (size_t)k0 * HDIM;  // 'b' term: B rows are k-indexed
    v8f amu = {}; v8f asg = {};
    #pragma unroll
    for (int kk = 0; kk < HDIM; kk += 32) {
      v16bf a = load_a(qmu + kk, HDIM, lane);
      v16bf b = load_b(kmu + kk, HDIM, lane);
      amu = wmma_bf16(a, b, amu);                        // mu_q @ mu_k^T
      a = load_a(qa + kk, HDIM, lane);
      b = load_b(ksg + kk, HDIM, lane);
      asg = wmma_bf16(a, b, asg);                        // (mu_q^2 + sg_q) @ sg_k^T
      a = load_a(km2 + kk, HDIM, lane);
      b = load_b(qsg + kk, HDIM, lane);
      asg = wmma_bf16(a, b, asg);                        // mu_k^2[q] @ sg_q[k]^T (index-swapped b)
    }
    int n = lane & 15, rb = (lane >> 4) * 8;
    #pragma unroll
    for (int r = 0; r < 8; ++r) {
      smu[(rb + r) * SP + k0 + n] = amu[r];
      ssg[(rb + r) * SP + k0 + n] = asg[r] * (1.0f / (float)HDIM);  // sigma_score / d
    }
  }
  __syncthreads();

  // ---- softmax + Sigma_weights + bf16 A-operand staging (8 threads per row) ----
  int row = tid >> 3, sub = tid & 7;
  float* srow = smu + row * SP;
  float* grow = ssg + row * SP;
  float mx = -3.0e38f;
  for (int j = sub; j < SEQ; j += 8) mx = fmaxf(mx, srow[j]);
  red[tid] = mx; __syncthreads();
  mx = red[row * 8];
  #pragma unroll
  for (int i = 1; i < 8; ++i) mx = fmaxf(mx, red[row * 8 + i]);
  __syncthreads();
  float sum = 0.0f;
  for (int j = sub; j < SEQ; j += 8) {
    float e = __expf((srow[j] - mx) * 0.125f);   // softmax of mu_score / sqrt(d)
    srow[j] = e; sum += e;
  }
  red[tid] = sum; __syncthreads();
  sum = 0.0f;
  #pragma unroll
  for (int i = 0; i < 8; ++i) sum += red[row * 8 + i];
  float inv = 1.0f / sum;
  for (int j = sub; j < SEQ; j += 8) {
    float w = srow[j] * inv;                       // mu_weights
    float g = w - w * w;
    float sw = g * g * grow[j] * (1.0f / (float)SEQ); // Sigma_weights = grad*(sig/d)/wdim
    int o = row * SPB + j;
    bW[o]  = (bf16)w;
    bW2[o] = (bf16)(w * w);
    bSw[o] = (bf16)sw;
  }
  __syncthreads();

  // ---- phase 2: outputs (each wave owns one 16-wide d-tile, K=1024) ----
  // Pure ds_load fragments + WMMA; all operands pre-converted to bf16.
  int n0 = wave * 16;
  const bf16* vmuT = hVmuT + hoff + (size_t)n0 * SEQ;
  const bf16* vsgT = hVsgT + hoff + (size_t)n0 * SEQ;
  const bf16* vxT  = hVxT  + hoff + (size_t)n0 * SEQ;
  v8f accmu = {}; v8f accsg = {};
  for (int k0 = 0; k0 < SEQ; k0 += 32) {
    v16bf aw  = load_a(bW  + k0, SPB, lane);             // W
    v16bf aw2 = load_a(bW2 + k0, SPB, lane);             // W^2
    v16bf asw = load_a(bSw + k0, SPB, lane);             // Sigma_weights
    v16bf bv  = load_b(vmuT + k0, SEQ, lane);
    v16bf bs  = load_b(vsgT + k0, SEQ, lane);
    v16bf bx  = load_b(vxT  + k0, SEQ, lane);
    accmu = wmma_bf16(aw,  bv, accmu);                   // W @ mu_v
    accsg = wmma_bf16(aw2, bs, accsg);                   // W^2 @ sg_v
    accsg = wmma_bf16(asw, bx, accsg);                   // Sw @ (mu_v^2 + sg_v)
  }
  int b = bh / HEADS, h = bh - b * HEADS;
  int n = lane & 15, rb = (lane >> 4) * 8;
  int col = h * HDIM + n0 + n;
  #pragma unroll
  for (int r = 0; r < 8; ++r) {
    int q = q0 + rb + r;
    size_t o = ((size_t)(b * SEQ + q)) * E_DIM + col;    // heads merged back to [B,S,E]
    Cmu[o] = (bf16)accmu[r];
    Csg[o] = (bf16)softplusf(accsg[r] * (1.0f / (float)SEQ)); // /wdim folded for all 3 terms
  }
}

// ---------------- KL reduction (single block) ----------------

__global__ void kl_kernel(const float* w0, const float* s0, const float* w1, const float* s1,
                          const float* w2, const float* s2, const float* w3, const float* s3,
                          float* __restrict__ out) {
  __shared__ float sred[256];
  const float* W[4]  = { w0, w1, w2, w3 };
  const float* Sg[4] = { s0, s1, s2, s3 };
  int tid = threadIdx.x;
  float total = 0.0f;
  for (int i = 0; i < 4; ++i) {
    float sm = 0.0f;
    for (int j = tid; j < E_DIM * E_DIM; j += 256) { float v = W[i][j]; sm += v * v; }
    sred[tid] = sm; __syncthreads();
    for (int o = 128; o; o >>= 1) { if (tid < o) sred[tid] += sred[tid + o]; __syncthreads(); }
    float sumMu = sred[0]; __syncthreads();
    float sv = 0.0f;
    for (int j = tid; j < E_DIM; j += 256) { float lv = Sg[i][j]; sv += -lv + softplusf(lv) * 100.0f; }
    sred[tid] = sv; __syncthreads();
    for (int o = 128; o; o >>= 1) { if (tid < o) sred[tid] += sred[tid + o]; __syncthreads(); }
    float sumSig = sred[0]; __syncthreads();
    total += 0.5f * (logf(0.01f) - 1.0f + sumSig / (float)E_DIM +
                     sumMu * 100.0f / ((float)E_DIM * (float)E_DIM));
  }
  if (tid == 0) out[0] = total;
}

// ---------------- host launch ----------------

extern "C" void kernel_launch(void* const* d_in, const int* in_sizes, int n_in,
                              void* d_out, int out_size, void* d_ws, size_t ws_size,
                              hipStream_t stream) {
  (void)in_sizes; (void)n_in; (void)out_size; (void)ws_size;
  const float* mu_in = (const float*)d_in[0];
  const float* sg_in = (const float*)d_in[1];
  const float* wmu[4] = { (const float*)d_in[2], (const float*)d_in[4],
                          (const float*)d_in[6], (const float*)d_in[8] };
  const float* wsg[4] = { (const float*)d_in[3], (const float*)d_in[5],
                          (const float*)d_in[7], (const float*)d_in[9] };

  char* base = (char*)d_ws;
  size_t cur = 0;
  auto alloc = [&](size_t bytes) -> void* {
    void* r = base + cur;
    cur = (cur + bytes + 255) & ~(size_t)255;
    return r;
  };
  const size_t NBF = (size_t)ROWS * E_DIM * sizeof(bf16);
  bf16* Xmu = (bf16*)alloc(NBF);  bf16* Xsg = (bf16*)alloc(NBF);
  float* xx = (float*)alloc(ROWS * sizeof(float));
  float* ss = (float*)alloc(ROWS * sizeof(float));
  bf16* wt[4]; bf16* w2t[4]; float* spw[4];
  for (int i = 0; i < 4; ++i) {
    wt[i]  = (bf16*)alloc((size_t)E_DIM * E_DIM * sizeof(bf16));
    w2t[i] = (bf16*)alloc((size_t)E_DIM * E_DIM * sizeof(bf16));
    spw[i] = (float*)alloc(E_DIM * sizeof(float));
  }
  bf16* Qmu = (bf16*)alloc(NBF); bf16* Qsg = (bf16*)alloc(NBF);
  bf16* Kmu = (bf16*)alloc(NBF); bf16* Ksg = (bf16*)alloc(NBF);
  bf16* Vmu = (bf16*)alloc(NBF); bf16* Vsg = (bf16*)alloc(NBF);
  bf16* hQmu = (bf16*)alloc(NBF); bf16* hQa  = (bf16*)alloc(NBF); bf16* hQsg = (bf16*)alloc(NBF);
  bf16* hKmu = (bf16*)alloc(NBF); bf16* hKmu2 = (bf16*)alloc(NBF); bf16* hKsg = (bf16*)alloc(NBF);
  bf16* hVmuT = (bf16*)alloc(NBF); bf16* hVsgT = (bf16*)alloc(NBF); bf16* hVxT = (bf16*)alloc(NBF);
  bf16* Cmu = (bf16*)alloc(NBF); bf16* Csg = (bf16*)alloc(NBF);
  float* cxx = (float*)alloc(ROWS * sizeof(float));
  float* css = (float*)alloc(ROWS * sizeof(float));

  float* out = (float*)d_out;
  const size_t MU_N = (size_t)ROWS * E_DIM;

  dim3 b256(256);
  for (int i = 0; i < 4; ++i)
    prep_weights_kernel<<<dim3((E_DIM * E_DIM + 255) / 256), b256, 0, stream>>>(
        wmu[i], wsg[i], wt[i], w2t[i], spw[i]);
  prep_inputs_kernel<<<dim3(ROWS), b256, 0, stream>>>(mu_in, sg_in, Xmu, Xsg, xx, ss);

  dim3 ggrid(E_DIM / 16, ROWS / 128);
  vdp_linear_kernel<<<ggrid, b256, 0, stream>>>(Xmu, Xsg, wt[0], w2t[0], spw[0], xx, ss,
                                                Qmu, Qsg, nullptr, nullptr, 0);
  vdp_linear_kernel<<<ggrid, b256, 0, stream>>>(Xmu, Xsg, wt[1], w2t[1], spw[1], xx, ss,
                                                Kmu, Ksg, nullptr, nullptr, 0);
  vdp_linear_kernel<<<ggrid, b256, 0, stream>>>(Xmu, Xsg, wt[2], w2t[2], spw[2], xx, ss,
                                                Vmu, Vsg, nullptr, nullptr, 0);

  pack_heads_kernel<<<dim3((unsigned)(HELEMS / 256)), b256, 0, stream>>>(
      Qmu, Qsg, Kmu, Ksg, Vmu, Vsg,
      hQmu, hQa, hQsg, hKmu, hKmu2, hKsg, hVmuT, hVsgT, hVxT);

  // f32 scores (2x16xSP) + red + three bf16 A-operand tiles (3x16xSPB)
  size_t shmem = (size_t)(32 * SP + 128) * sizeof(float) + (size_t)(3 * 16 * SPB) * sizeof(bf16);
  attention_kernel<<<dim3(BH, SEQ / 16), dim3(128), shmem, stream>>>(
      hQmu, hQa, hQsg, hKmu, hKmu2, hKsg, hVmuT, hVsgT, hVxT, Cmu, Csg);

  cat_stats_kernel<<<dim3(ROWS), b256, 0, stream>>>(Cmu, Csg, cxx, css);
  vdp_linear_kernel<<<ggrid, b256, 0, stream>>>(Cmu, Csg, wt[3], w2t[3], spw[3], cxx, css,
                                                nullptr, nullptr, out, out + MU_N, 1);
  kl_kernel<<<dim3(1), b256, 0, stream>>>(wmu[0], wsg[0], wmu[1], wsg[1],
                                          wmu[2], wsg[2], wmu[3], wsg[3], out + 2 * MU_N);
}
